// MoMKE_91233695301751
// MI455X (gfx1250) — matrix operand order repaired and checked by reading, not compiled, not run on hardware
//
#include <hip/hip_runtime.h>
#include <cstring>

// ---------------------------------------------------------------------------
// Model dimensions (from the reference)
// ---------------------------------------------------------------------------
constexpr int BATCH = 16, SEQ = 128;
constexpr int ADIMS = 512, TDIMS = 768, VDIMS = 1024;
constexpr int DE = 512, NMOD = 3, NTOK = NMOD * SEQ;       // 384 tokens / batch
constexpr long long MTOK = (long long)BATCH * NTOK;        // 6144 rows
constexpr int NHEAD = 8, HD = 64, NEXP = 6, FF = 512, DEPTH = 4, NCLS = 6;
constexpr int D3 = 3 * DE;                                 // 1536

// GEMM tiling: 256 threads = 8 wave32; wave grid 4x2, each wave 32x32 (2x2 WMMA)
constexpr int BM = 128, BN = 64, BKT = 32;
constexpr int LDK = BKT + 8;   // padded K stride for LDS tiles (80B rows: 16B-aligned)

typedef __attribute__((ext_vector_type(16))) __bf16 v16bf;
typedef __attribute__((ext_vector_type(8)))  float  v8f;

// ---------------------------------------------------------------------------
// Device helpers
// ---------------------------------------------------------------------------
__device__ __forceinline__ unsigned short f32_to_bf16(float f) {
  unsigned int u = __float_as_uint(f);
  u += 0x7FFFu + ((u >> 16) & 1u);          // round-to-nearest-even
  return (unsigned short)(u >> 16);
}
__device__ __forceinline__ float bf16_to_f32(unsigned short h) {
  return __uint_as_float(((unsigned int)h) << 16);
}
__device__ __forceinline__ float gelu_f(float x) {
  float x3 = x * x * x;
  return 0.5f * x * (1.0f + tanhf(0.7978845608028654f * (x + 0.044715f * x3)));
}

// ---------------------------------------------------------------------------
// Generic batched WMMA GEMM:  C[z] = act(alpha * A[z] @ B[z] + bias[z]) (+resid)
// A: (M,K) bf16 row-major.  B: (K,N) bf16 row-major, or (N,K) if TRB.
// Two-level batch index: z1 = z / nb2, z2 = z % nb2 (for (batch,head) batching).
// K must be a multiple of BKT (true for every GEMM in this model).
// EDGE=false: M%BM==0, N%BN==0, lda%4==0, ldb%4==0 guaranteed by the host ->
//             no guard code at all in the instantiation (hot path).
// ---------------------------------------------------------------------------
struct GemmP {
  const unsigned short* A;  long long aS1, aS2;  int lda;
  const unsigned short* B;  long long bS1, bS2;  int ldb;
  const float* bias;        long long biS1;
  const float* resid;       int ldr;
  float* Cf;                long long cfS1, cfS2; int ldcf;
  unsigned short* Cb;       long long cbS1, cbS2; int ldcb;
  int M, N, K, nb2;
  float alpha;
};

template<int ACT, bool TRB, bool HASBIAS, bool HASRESID, bool OUTF, bool OUTB, bool EDGE>
__global__ __launch_bounds__(256) void k_gemm(GemmP p) {
  static_assert(BKT == 32 && BN == 64 && BM == 128, "tile index math");
  // Double-buffered tiles; both A and B are stored K-contiguous so the
  // WMMA fragment gathers are ds_load_b128 runs.
  __shared__ unsigned short As[2][BM][LDK];   // [m][k]
  __shared__ unsigned short Bs[2][BN][LDK];   // [n][k]  (n-major!)

  const int z  = blockIdx.z;
  const int z1 = z / p.nb2, z2 = z % p.nb2;
  const unsigned short* Ap = p.A + z1 * p.aS1 + z2 * p.aS2;
  const unsigned short* Bp = p.B + z1 * p.bS1 + z2 * p.bS2;

  const int row0 = blockIdx.y * BM;
  const int col0 = blockIdx.x * BN;
  const int wave = threadIdx.x >> 5, lane = threadIdx.x & 31;
  const int wm = wave >> 1, wn = wave & 1;          // 4 x 2 wave grid
  const int half = lane >> 4, r = lane & 15;

  // ---- tile staging ----
  auto stage = [&](int b, int kk) {
    if (!EDGE) {
      // Guard-free, vectorized (host guarantees alignment & tile-multiple M/N).
      for (int t = threadIdx.x; t < BM * (BKT / 4); t += 256) {
        int m = t >> 3, kc = (t & 7) << 2;
        ushort4 v = *(const ushort4*)(Ap + (long long)(row0 + m) * p.lda + kk + kc);
        *(ushort4*)(&As[b][m][kc]) = v;
      }
      if (TRB) {
        // B rows are K-contiguous in memory: vector load + vector LDS store
        for (int t = threadIdx.x; t < BN * (BKT / 4); t += 256) {
          int n = t >> 3, kc = (t & 7) << 2;
          ushort4 v = *(const ushort4*)(Bp + (long long)(col0 + n) * p.ldb + kk + kc);
          *(ushort4*)(&Bs[b][n][kc]) = v;
        }
      } else {
        // B is N-contiguous in memory: vector load along n, scatter into LDS
        for (int t = threadIdx.x; t < BKT * (BN / 4); t += 256) {
          int k = t >> 4, nc = (t & 15) << 2;
          ushort4 v = *(const ushort4*)(Bp + (long long)(kk + k) * p.ldb + col0 + nc);
          Bs[b][nc + 0][k] = v.x;
          Bs[b][nc + 1][k] = v.y;
          Bs[b][nc + 2][k] = v.z;
          Bs[b][nc + 3][k] = v.w;
        }
      }
    } else {
      // Ragged shapes (head M=16, logits N=6): guarded scalar staging.
      for (int t = threadIdx.x; t < BM * BKT; t += 256) {
        int m = t >> 5, k = t & 31;
        int gm = row0 + m;
        As[b][m][k] = (gm < p.M) ? Ap[(long long)gm * p.lda + kk + k] : (unsigned short)0;
      }
      for (int t = threadIdx.x; t < BN * BKT; t += 256) {
        int n = t >> 5, k = t & 31;
        int gn = col0 + n, gk = kk + k;
        unsigned short v = 0;
        if (gn < p.N)
          v = TRB ? Bp[(long long)gn * p.ldb + gk] : Bp[(long long)gk * p.ldb + gn];
        Bs[b][n][k] = v;
      }
    }
  };

  v8f acc[2][2] = {};

  int buf = 0;
  stage(0, 0);
  __syncthreads();

  for (int k0 = 0; k0 < p.K; k0 += BKT) {
    const int nk = k0 + BKT;
    if (nk < p.K) stage(buf ^ 1, nk);

    // prefetch the K+2 tile into the cache hierarchy while WMMAs run
    if (!EDGE && nk + BKT < p.K) {
      if (threadIdx.x < BM) {
        __builtin_prefetch(Ap + (long long)(row0 + threadIdx.x) * p.lda + nk + BKT, 0, 1);
      } else if (TRB) {
        if (threadIdx.x - BM < BN)
          __builtin_prefetch(Bp + (long long)(col0 + threadIdx.x - BM) * p.ldb + nk + BKT, 0, 1);
      } else {
        if (threadIdx.x - BM < BKT)
          __builtin_prefetch(Bp + (long long)(nk + BKT + threadIdx.x - BM) * p.ldb + col0, 0, 1);
      }
    }

    // ---- pack fragments per ISA 7.12.2 layouts (contiguous 16B LDS runs) ----
    union Frag { v16bf v; unsigned short u[16]; };
    Frag af[2], bf[2];
#pragma unroll
    for (int i = 0; i < 2; i++) {
      int m = wm * 32 + i * 16 + r;
#pragma unroll
      for (int j = 0; j < 16; j++) {
        int kk = ((j >> 3) << 4) + (half << 3) + (j & 7);  // K 0-7/16-23 vs 8-15/24-31
        af[i].u[j] = As[buf][m][kk];
      }
    }
#pragma unroll
    for (int i = 0; i < 2; i++) {
      int n = wn * 32 + i * 16 + r;
#pragma unroll
      for (int j = 0; j < 16; j++) {
        int kk = ((j >> 3) << 4) + (half << 3) + (j & 7);  // K 0-7/16-23 vs 8-15/24-31
        bf[i].u[j] = Bs[buf][n][kk];
      }
    }
#pragma unroll
    for (int i = 0; i < 2; i++)
#pragma unroll
      for (int jn = 0; jn < 2; jn++)
        acc[i][jn] = __builtin_amdgcn_wmma_f32_16x16x32_bf16(
            false, af[i].v, false, bf[jn].v, (short)0, acc[i][jn], false, false);

    __syncthreads();      // one barrier per K-step (covers staging + reads)
    buf ^= 1;
  }

  // -------- epilogue --------
  const float* biasp = HASBIAS ? (p.bias + z1 * p.biS1) : nullptr;
  const long long cfz = z1 * p.cfS1 + z2 * p.cfS2;
  const long long cbz = z1 * p.cbS1 + z2 * p.cbS2;
#pragma unroll
  for (int i = 0; i < 2; i++)
#pragma unroll
    for (int jn = 0; jn < 2; jn++)
#pragma unroll
      for (int e = 0; e < 8; e++) {
        int gm = row0 + wm * 32 + i * 16 + half * 8 + e;   // C/D: M = half*8 + vgpr
        int gn = col0 + wn * 32 + jn * 16 + r;             // N = lane % 16
        if (!EDGE || (gm < p.M && gn < p.N)) {
          float v = acc[i][jn][e] * p.alpha;
          if (HASBIAS) v += biasp[gn];
          if (ACT == 1) v = gelu_f(v);
          else if (ACT == 2) v = fmaxf(v, 0.0f);
          if (HASRESID) v += p.resid[(long long)gm * p.ldr + gn];
          if (OUTF) p.Cf[cfz + (long long)gm * p.ldcf + gn] = v;
          if (OUTB) p.Cb[cbz + (long long)gm * p.ldcb + gn] = f32_to_bf16(v);
        }
      }
}

// ---------------------------------------------------------------------------
// Small support kernels
// ---------------------------------------------------------------------------
__global__ void k_f32_to_bf16(const float* __restrict__ in, unsigned short* __restrict__ out,
                              long long n) {
  long long i = (long long)blockIdx.x * 256 + threadIdx.x;
  long long stride = (long long)gridDim.x * 256;
  for (; i < n; i += stride) out[i] = f32_to_bf16(in[i]);
}

// LayerNorm over width (one block of 128 threads per row), bf16 output
__global__ void k_layernorm(const float* __restrict__ x, const float* __restrict__ g,
                            const float* __restrict__ b, unsigned short* __restrict__ out,
                            int width) {
  long long row = blockIdx.x;
  const float* xr = x + row * width;
  __shared__ float sm[128], sv[128];
  float s = 0.f, q = 0.f;
  for (int i = threadIdx.x; i < width; i += 128) { float v = xr[i]; s += v; q += v * v; }
  sm[threadIdx.x] = s; sv[threadIdx.x] = q; __syncthreads();
  for (int st = 64; st > 0; st >>= 1) {
    if (threadIdx.x < st) { sm[threadIdx.x] += sm[threadIdx.x + st]; sv[threadIdx.x] += sv[threadIdx.x + st]; }
    __syncthreads();
  }
  float mean = sm[0] / width;
  float var  = sv[0] / width - mean * mean;
  float rstd = rsqrtf(var + 1e-5f);
  for (int i = threadIdx.x; i < width; i += 128) {
    float v = (xr[i] - mean) * rstd * g[i] + b[i];
    out[row * width + i] = f32_to_bf16(v);
  }
}

// Row softmax, fp32 in -> bf16 out (one block of 128 threads per row)
__global__ void k_softmax(const float* __restrict__ s, unsigned short* __restrict__ out,
                          int width) {
  long long row = blockIdx.x;
  const float* sr = s + row * width;
  __shared__ float red[128];
  float mx = -1e30f;
  for (int i = threadIdx.x; i < width; i += 128) mx = fmaxf(mx, sr[i]);
  red[threadIdx.x] = mx; __syncthreads();
  for (int st = 64; st > 0; st >>= 1) {
    if (threadIdx.x < st) red[threadIdx.x] = fmaxf(red[threadIdx.x], red[threadIdx.x + st]);
    __syncthreads();
  }
  mx = red[0]; __syncthreads();
  float sum = 0.f;
  for (int i = threadIdx.x; i < width; i += 128) sum += expf(sr[i] - mx);
  red[threadIdx.x] = sum; __syncthreads();
  for (int st = 64; st > 0; st >>= 1) {
    if (threadIdx.x < st) red[threadIdx.x] += red[threadIdx.x + st];
    __syncthreads();
  }
  float inv = 1.0f / red[0];
  for (int i = threadIdx.x; i < width; i += 128)
    out[row * width + i] = f32_to_bf16(expf(sr[i] - mx) * inv);
}

// (3,B,S,DE) fp32 -> (B,3S,DE) fp32 residual stream
__global__ void k_permute_x(const float* __restrict__ xm, float* __restrict__ x) {
  long long n = 3LL * BATCH * SEQ * DE;
  for (long long i = (long long)blockIdx.x * 256 + threadIdx.x; i < n;
       i += (long long)gridDim.x * 256) {
    long long d = i % DE, rem = i / DE;
    long long s = rem % SEQ; rem /= SEQ;
    long long b = rem % BATCH;
    long long m = rem / BATCH;
    x[(((b * 3 + m) * SEQ + s) * DE) + d] = xm[i];
  }
}

// top-2 softmax gates; rlog is (3,B,S,E); gates is (B,3S,E) (token order of x)
__global__ void k_gates(const float* __restrict__ rlog, float* __restrict__ gates) {
  int t = blockIdx.x * 256 + threadIdx.x;
  int ntok = 3 * BATCH * SEQ;
  if (t >= ntok) return;
  float v[NEXP];
  for (int e = 0; e < NEXP; e++) v[e] = rlog[(long long)t * NEXP + e];
  int i1 = 0;
  for (int e = 1; e < NEXP; e++) if (v[e] > v[i1]) i1 = e;
  int i2 = -1;
  for (int e = 0; e < NEXP; e++) { if (e == i1) continue; if (i2 < 0 || v[e] > v[i2]) i2 = e; }
  float e2 = expf(v[i2] - v[i1]);
  float g1 = 1.0f / (1.0f + e2), g2 = e2 / (1.0f + e2);
  int m = t / (BATCH * SEQ), rem = t % (BATCH * SEQ), b = rem / SEQ, s = rem % SEQ;
  long long tok = ((long long)b * 3 + m) * SEQ + s;
  for (int e = 0; e < NEXP; e++)
    gates[tok * NEXP + e] = (e == i1) ? g1 : (e == i2) ? g2 : 0.0f;
}

// x += shared + sum_e gates[tok,e] * eo[e,tok,:]
__global__ void k_combine(float* __restrict__ x, const float* __restrict__ shr,
                          const float* __restrict__ eo, const float* __restrict__ gates) {
  long long n = MTOK * DE;
  for (long long i = (long long)blockIdx.x * 256 + threadIdx.x; i < n;
       i += (long long)gridDim.x * 256) {
    long long tok = i / DE;
    float acc = x[i] + shr[i];
#pragma unroll
    for (int e = 0; e < NEXP; e++) acc += gates[tok * NEXP + e] * eo[(long long)e * n + i];
    x[i] = acc;
  }
}

// mean over S per (b, modality, d) -> pooled (B, 1536) bf16
__global__ void k_pool(const unsigned short* __restrict__ xf, unsigned short* __restrict__ pooled) {
  int idx = blockIdx.x * 256 + threadIdx.x;
  int n = BATCH * 3 * DE;
  if (idx >= n) return;
  int d = idx % DE, rem = idx / DE, m = rem % 3, b = rem / 3;
  float s = 0.f;
  for (int ss = 0; ss < SEQ; ss++)
    s += bf16_to_f32(xf[(((long long)b * 3 + m) * SEQ + ss) * DE + d]);
  pooled[(long long)b * D3 + m * DE + d] = f32_to_bf16(s * (1.0f / SEQ));
}

// ---------------------------------------------------------------------------
// Host orchestration
// ---------------------------------------------------------------------------
static inline GemmP mkgemm() { GemmP p; memset(&p, 0, sizeof(p)); p.nb2 = 1; p.alpha = 1.0f; return p; }
static inline dim3 gemm_grid(const GemmP& p, int nb) {
  return dim3((p.N + BN - 1) / BN, (p.M + BM - 1) / BM, nb);
}
static inline int gs_blocks(long long n) {
  long long b = (n + 255) / 256; if (b > 8192) b = 8192; return (int)b;
}

extern "C" void kernel_launch(void* const* d_in, const int* in_sizes, int n_in,
                              void* d_out, int out_size, void* d_ws, size_t ws_size,
                              hipStream_t stream) {
  // inputs in setup_inputs() order
  const float* audio = (const float*)d_in[0];
  const float* text  = (const float*)d_in[1];
  const float* visual= (const float*)d_in[2];
  const float* Wa = (const float*)d_in[3];  const float* ba = (const float*)d_in[4];
  const float* Wt = (const float*)d_in[5];  const float* bt = (const float*)d_in[6];
  const float* Wv = (const float*)d_in[7];  const float* bv = (const float*)d_in[8];
  const float* Wr1= (const float*)d_in[9];  const float* br1= (const float*)d_in[10];
  const float* Wr2= (const float*)d_in[11]; const float* br2= (const float*)d_in[12];
  const float* ln1s=(const float*)d_in[13]; const float* ln1b=(const float*)d_in[14];
  const float* Wqkv=(const float*)d_in[15]; const float* bqkv=(const float*)d_in[16];
  const float* Wo = (const float*)d_in[17]; const float* bo = (const float*)d_in[18];
  const float* ln2s=(const float*)d_in[19]; const float* ln2b=(const float*)d_in[20];
  const float* Ws1= (const float*)d_in[21]; const float* bs1= (const float*)d_in[22];
  const float* Ws2= (const float*)d_in[23]; const float* bs2= (const float*)d_in[24];
  const float* We1= (const float*)d_in[25]; const float* be1= (const float*)d_in[26];
  const float* We2= (const float*)d_in[27]; const float* be2= (const float*)d_in[28];
  const float* lnfs=(const float*)d_in[29]; const float* lnfb=(const float*)d_in[30];
  const float* Wp1= (const float*)d_in[31]; const float* bp1= (const float*)d_in[32];
  const float* Wh = (const float*)d_in[33]; const float* bh = (const float*)d_in[34];
  (void)in_sizes; (void)n_in; (void)out_size; (void)ws_size;

  // bump allocator over workspace
  char* base = (char*)d_ws; size_t off = 0;
  auto alloc = [&](size_t bytes) -> void* {
    size_t a = (off + 255) & ~(size_t)255; void* p = base + a; off = a + bytes; return p;
  };
  auto cvt = [&](const float* src, long long n) -> unsigned short* {
    unsigned short* dst = (unsigned short*)alloc((size_t)n * 2);
    k_f32_to_bf16<<<gs_blocks(n), 256, 0, stream>>>(src, dst, n);
    return dst;
  };

  // ---- bf16 copies of all GEMM operands (weights + raw inputs) ----
  unsigned short* wa_b  = cvt(Wa,  (long long)ADIMS * DE);
  unsigned short* wt_b  = cvt(Wt,  (long long)TDIMS * DE);
  unsigned short* wv_b  = cvt(Wv,  (long long)VDIMS * DE);
  unsigned short* wr1_b = cvt(Wr1, 3LL * DE * DE);
  unsigned short* wr2_b = cvt(Wr2, 3LL * DE * NEXP);
  unsigned short* wqkv_b= cvt(Wqkv,(long long)DEPTH * DE * D3);
  unsigned short* wo_b  = cvt(Wo,  (long long)DEPTH * DE * DE);
  unsigned short* ws1_b = cvt(Ws1, (long long)DEPTH * DE * FF);
  unsigned short* ws2_b = cvt(Ws2, (long long)DEPTH * FF * DE);
  unsigned short* we1_b = cvt(We1, (long long)DEPTH * NEXP * DE * FF);
  unsigned short* we2_b = cvt(We2, (long long)DEPTH * NEXP * FF * DE);
  unsigned short* wp1_b = cvt(Wp1, (long long)D3 * D3);
  unsigned short* wh_b  = cvt(Wh,  (long long)D3 * NCLS);
  unsigned short* a_b   = cvt(audio, (long long)BATCH * SEQ * ADIMS);
  unsigned short* t_b   = cvt(text,  (long long)BATCH * SEQ * TDIMS);
  unsigned short* v_b   = cvt(visual,(long long)BATCH * SEQ * VDIMS);

  // ---- activations ----
  const long long MS = (long long)BATCH * SEQ;              // 2048 rows / modality
  float*          xmods_f = (float*)alloc((size_t)(3 * MS * DE) * 4);  // also reused as "shared"
  unsigned short* xmods_b = (unsigned short*)alloc((size_t)(3 * MS * DE) * 2);
  float*          x       = (float*)alloc((size_t)(MTOK * DE) * 4);
  unsigned short* hbuf    = (unsigned short*)alloc((size_t)(MTOK * DE) * 2);
  unsigned short* qkv_b   = (unsigned short*)alloc((size_t)(MTOK * D3) * 2);
  float*          big_f   = (float*)alloc((size_t)(NEXP * MTOK * DE) * 4);  // scores / eo (same size)
  unsigned short* big_b   = (unsigned short*)alloc((size_t)(NEXP * MTOK * FF) * 2); // att / eh
  unsigned short* obuf    = (unsigned short*)alloc((size_t)(MTOK * DE) * 2); // attn O / shared-hidden
  unsigned short* rh_b    = (unsigned short*)alloc((size_t)(3 * MS * DE) * 2);
  float*          rlog    = (float*)alloc((size_t)(3 * MS * NEXP) * 4);
  float*          gates   = (float*)alloc((size_t)(MTOK * NEXP) * 4);
  unsigned short* pooled_b= (unsigned short*)alloc((size_t)BATCH * D3 * 2);
  unsigned short* fused_b = (unsigned short*)alloc((size_t)BATCH * D3 * 2);

  // ---- 1) modality projections: x_mod = in @ W + b  (fp32 + bf16 outputs) ----
  {
    const unsigned short* ins[3] = { a_b, t_b, v_b };
    const unsigned short* ws[3]  = { wa_b, wt_b, wv_b };
    const float* bs[3] = { ba, bt, bv };
    const int ks[3] = { ADIMS, TDIMS, VDIMS };
    for (int m = 0; m < 3; m++) {
      GemmP p = mkgemm();
      p.A = ins[m]; p.lda = ks[m];
      p.B = ws[m];  p.ldb = DE;
      p.bias = bs[m];
      p.Cf = xmods_f + (long long)m * MS * DE; p.ldcf = DE;
      p.Cb = xmods_b + (long long)m * MS * DE; p.ldcb = DE;
      p.M = (int)MS; p.N = DE; p.K = ks[m];
      hipLaunchKernelGGL((k_gemm<0,false,true,false,true,true,false>), gemm_grid(p,1), dim3(256), 0, stream, p);
    }
  }
  // residual stream x = permute (3,B,S,DE) -> (B,3S,DE)
  k_permute_x<<<gs_blocks(3 * MS * DE), 256, 0, stream>>>(xmods_f, x);

  // ---- 2) router: rh = gelu(x_mod @ Wr1 + br1); rlog = rh @ Wr2 + br2 ----
  {
    GemmP p = mkgemm();
    p.A = xmods_b; p.aS1 = MS * DE; p.lda = DE;
    p.B = wr1_b;   p.bS1 = (long long)DE * DE; p.ldb = DE;
    p.bias = br1;  p.biS1 = DE;
    p.Cb = rh_b;   p.cbS1 = MS * DE; p.ldcb = DE;
    p.M = (int)MS; p.N = DE; p.K = DE;
    hipLaunchKernelGGL((k_gemm<1,false,true,false,false,true,false>), gemm_grid(p,3), dim3(256), 0, stream, p);
  }
  {
    GemmP p = mkgemm();
    p.A = rh_b;  p.aS1 = MS * DE; p.lda = DE;
    p.B = wr2_b; p.bS1 = (long long)DE * NEXP; p.ldb = NEXP;
    p.bias = br2; p.biS1 = NEXP;
    p.Cf = rlog;  p.cfS1 = MS * NEXP; p.ldcf = NEXP;
    p.M = (int)MS; p.N = NEXP; p.K = DE;
    hipLaunchKernelGGL((k_gemm<0,false,true,false,true,false,true>), gemm_grid(p,3), dim3(256), 0, stream, p);
  }
  k_gates<<<(3 * (int)MS + 255) / 256, 256, 0, stream>>>(rlog, gates);

  // ---- 3) transformer layers ----
  const float attn_scale = 0.125f;  // 1/sqrt(HD=64)
  for (int l = 0; l < DEPTH; l++) {
    // ln1
    k_layernorm<<<(int)MTOK, 128, 0, stream>>>(x, ln1s + (long long)l * DE, ln1b + (long long)l * DE, hbuf, DE);
    // qkv
    {
      GemmP p = mkgemm();
      p.A = hbuf; p.lda = DE;
      p.B = wqkv_b + (long long)l * DE * D3; p.ldb = D3;
      p.bias = bqkv + (long long)l * D3;
      p.Cb = qkv_b; p.ldcb = D3;
      p.M = (int)MTOK; p.N = D3; p.K = DE;
      hipLaunchKernelGGL((k_gemm<0,false,true,false,false,true,false>), gemm_grid(p,1), dim3(256), 0, stream, p);
    }
    // scores = (Q @ K^T) * scale : batched over (b,h), B transposed
    {
      GemmP p = mkgemm();
      p.nb2 = NHEAD;
      p.A = qkv_b;      p.aS1 = (long long)NTOK * D3; p.aS2 = HD; p.lda = D3;
      p.B = qkv_b + DE; p.bS1 = (long long)NTOK * D3; p.bS2 = HD; p.ldb = D3;
      p.Cf = big_f;     p.cfS1 = (long long)NHEAD * NTOK * NTOK; p.cfS2 = (long long)NTOK * NTOK; p.ldcf = NTOK;
      p.M = NTOK; p.N = NTOK; p.K = HD; p.alpha = attn_scale;
      hipLaunchKernelGGL((k_gemm<0,true,false,false,true,false,false>), gemm_grid(p, BATCH * NHEAD), dim3(256), 0, stream, p);
    }
    // softmax rows -> bf16 probabilities
    k_softmax<<<BATCH * NHEAD * NTOK, 128, 0, stream>>>(big_f, big_b, NTOK);
    // O = att @ V : batched over (b,h)
    {
      GemmP p = mkgemm();
      p.nb2 = NHEAD;
      p.A = big_b;          p.aS1 = (long long)NHEAD * NTOK * NTOK; p.aS2 = (long long)NTOK * NTOK; p.lda = NTOK;
      p.B = qkv_b + 2 * DE; p.bS1 = (long long)NTOK * D3; p.bS2 = HD; p.ldb = D3;
      p.Cb = obuf;          p.cbS1 = (long long)NTOK * DE; p.cbS2 = HD; p.ldcb = DE;
      p.M = NTOK; p.N = HD; p.K = NTOK;
      hipLaunchKernelGGL((k_gemm<0,false,false,false,false,true,false>), gemm_grid(p, BATCH * NHEAD), dim3(256), 0, stream, p);
    }
    // x = x + O @ Wo + bo
    {
      GemmP p = mkgemm();
      p.A = obuf; p.lda = DE;
      p.B = wo_b + (long long)l * DE * DE; p.ldb = DE;
      p.bias = bo + (long long)l * DE;
      p.resid = x; p.ldr = DE;
      p.Cf = x; p.ldcf = DE;
      p.M = (int)MTOK; p.N = DE; p.K = DE;
      hipLaunchKernelGGL((k_gemm<0,false,true,true,true,false,false>), gemm_grid(p,1), dim3(256), 0, stream, p);
    }
    // ln2
    k_layernorm<<<(int)MTOK, 128, 0, stream>>>(x, ln2s + (long long)l * DE, ln2b + (long long)l * DE, hbuf, DE);
    // shared FFN hidden: gelu(h @ Ws1 + bs1)
    {
      GemmP p = mkgemm();
      p.A = hbuf; p.lda = DE;
      p.B = ws1_b + (long long)l * DE * FF; p.ldb = FF;
      p.bias = bs1 + (long long)l * FF;
      p.Cb = obuf; p.ldcb = FF;
      p.M = (int)MTOK; p.N = FF; p.K = DE;
      hipLaunchKernelGGL((k_gemm<1,false,true,false,false,true,false>), gemm_grid(p,1), dim3(256), 0, stream, p);
    }
    // shared FFN out (fp32, combined later)
    {
      GemmP p = mkgemm();
      p.A = obuf; p.lda = FF;
      p.B = ws2_b + (long long)l * FF * DE; p.ldb = DE;
      p.bias = bs2 + (long long)l * DE;
      p.Cf = xmods_f; p.ldcf = DE;     // reuse buffer as "shared" output
      p.M = (int)MTOK; p.N = DE; p.K = FF;
      hipLaunchKernelGGL((k_gemm<0,false,true,false,true,false,false>), gemm_grid(p,1), dim3(256), 0, stream, p);
    }
    // experts hidden: batched over e, shared A
    {
      GemmP p = mkgemm();
      p.A = hbuf; p.aS1 = 0; p.lda = DE;
      p.B = we1_b + (long long)l * NEXP * DE * FF; p.bS1 = (long long)DE * FF; p.ldb = FF;
      p.bias = be1 + (long long)l * NEXP * FF; p.biS1 = FF;
      p.Cb = big_b; p.cbS1 = MTOK * FF; p.ldcb = FF;
      p.M = (int)MTOK; p.N = FF; p.K = DE;
      hipLaunchKernelGGL((k_gemm<1,false,true,false,false,true,false>), gemm_grid(p, NEXP), dim3(256), 0, stream, p);
    }
    // experts out (fp32): batched over e
    {
      GemmP p = mkgemm();
      p.A = big_b; p.aS1 = MTOK * FF; p.lda = FF;
      p.B = we2_b + (long long)l * NEXP * FF * DE; p.bS1 = (long long)FF * DE; p.ldb = DE;
      p.bias = be2 + (long long)l * NEXP * DE; p.biS1 = DE;
      p.Cf = big_f; p.cfS1 = MTOK * DE; p.ldcf = DE;
      p.M = (int)MTOK; p.N = DE; p.K = FF;
      hipLaunchKernelGGL((k_gemm<0,false,true,false,true,false,false>), gemm_grid(p, NEXP), dim3(256), 0, stream, p);
    }
    // x += shared + gated expert sum
    k_combine<<<gs_blocks(MTOK * DE), 256, 0, stream>>>(x, xmods_f, big_f, gates);
  }

  // ---- 4) head ----
  k_layernorm<<<(int)MTOK, 128, 0, stream>>>(x, lnfs, lnfb, hbuf, DE);
  k_pool<<<(BATCH * 3 * DE + 255) / 256, 256, 0, stream>>>(hbuf, pooled_b);
  {
    GemmP p = mkgemm();
    p.A = pooled_b; p.lda = D3;
    p.B = wp1_b;    p.ldb = D3;
    p.bias = bp1;
    p.Cb = fused_b; p.ldcb = D3;
    p.M = BATCH; p.N = D3; p.K = D3;
    hipLaunchKernelGGL((k_gemm<2,false,true,false,false,true,true>), gemm_grid(p,1), dim3(256), 0, stream, p);
  }
  {
    GemmP p = mkgemm();
    p.A = fused_b; p.lda = D3;
    p.B = wh_b;    p.ldb = NCLS;
    p.bias = bh;
    p.Cf = (float*)d_out; p.ldcf = NCLS;
    p.M = BATCH; p.N = NCLS; p.K = D3;
    hipLaunchKernelGGL((k_gemm<0,false,true,false,true,false,true>), gemm_grid(p,1), dim3(256), 0, stream, p);
  }
}
